// GoldenSeedSelection_67817533604024
// MI455X (gfx1250) — matrix operand order repaired and checked by reading, not compiled, not run on hardware
//
#include <hip/hip_runtime.h>
#include <math.h>

// Problem constants (match reference)
constexpr int B = 8, Q = 2048, C = 1024, G = 128;
constexpr float EPS = 1e-12f;
constexpr float INF = 3.0e38f;

typedef __attribute__((ext_vector_type(8)))  float   v8f;
typedef __attribute__((ext_vector_type(16))) float   v16f;
typedef __attribute__((ext_vector_type(8)))  __bf16  v8bf;
typedef __attribute__((ext_vector_type(16))) __bf16  v16bf;

// ---------------------------------------------------------------------------
// Kernel 1: reciprocal row norms. One 256-thread block per row of C=1024.
// rnorm[row] = 1 / max(||src[row,:]||, 1e-12)
// ---------------------------------------------------------------------------
__global__ void rnorm_kernel(const float* __restrict__ src,
                             float* __restrict__ rnorm) {
  const int row = blockIdx.x;
  const float* s = src + (size_t)row * C;
  float ss = 0.f;
  for (int i = threadIdx.x; i < C; i += 256) {
    float x = s[i];
    ss += x * x;
  }
  __shared__ float red[256];
  red[threadIdx.x] = ss;
  __syncthreads();
  for (int off = 128; off > 0; off >>= 1) {
    if (threadIdx.x < off) red[threadIdx.x] += red[threadIdx.x + off];
    __syncthreads();
  }
  if (threadIdx.x == 0)
    rnorm[row] = 1.0f / fmaxf(sqrtf(red[0]), EPS);
}

// ---------------------------------------------------------------------------
// Kernel 2: cost[b,g,q] = 1 - rq[b,q]*rg[b,g]*<q[b,q,:], gt[b,g,:]>
// One wave per (batch, 16-wide q-tile); the wave sweeps ALL of G (8 tiles of
// 16) holding 8 f32 accumulators (64 VGPRs).  Per K-step of 32:
//   1) the B operand (q data) is loaded/converted ONCE,
//   2) all 8 A tiles are staged into DISTINCT register sets,
//   3) 8 independent v_wmma_f32_16x16x32_bf16 are issued back-to-back.
// Staging all A tiles before the WMMA burst puts >= a full K-step of
// instructions between any WMMA and the next write to its source VGPRs, so
// the 4-coexec WMMA->VALU WAR hazard (ISA 7.12.1) needs no v_nop padding.
// Per-lane operands use contiguous k order (k = 16*half + s) for both A and B,
// not the ISA table order; valid because A and B use the same order.
// ---------------------------------------------------------------------------
__global__ __launch_bounds__(256)
void cost_gemm_kernel(const float* __restrict__ qsig,
                      const float* __restrict__ gsig,
                      const float* __restrict__ rq,
                      const float* __restrict__ rg,
                      float* __restrict__ cost) {
  constexpr int QT = Q / 16;      // 128 q-tiles per batch
  constexpr int GTiles = G / 16;  // 8 g-tiles, all handled by this wave

  const int lane = threadIdx.x & 31;
  const int wave = threadIdx.x >> 5;
  const int job  = blockIdx.x * (blockDim.x >> 5) + wave;  // 0 .. B*QT-1
  const int qt = job % QT;
  const int b  = job / QT;

  const int half = lane >> 4;   // 0: lanes 0-15, 1: lanes 16-31
  const int m    = lane & 15;

  // B operand (q rows as columns): lane holds column N=m; K run
  // [k0 + half*16, k0 + half*16 + 16).
  const float* brow = qsig + ((size_t)b * Q + (size_t)(qt * 16 + m)) * C + half * 16;
  // A operand (gt rows): lane holds row M=m of g-tile g8; K run
  // [k0 + half*16, k0 + half*16 + 16).
  const float* arow0 = gsig + ((size_t)b * G + m) * C + half * 16;

  union AU { v16bf v; v8bf h[2]; };
  v8f acc[GTiles];
  #pragma unroll
  for (int g8 = 0; g8 < GTiles; ++g8) acc[g8] = (v8f){};

  for (int k0 = 0; k0 < C; k0 += 32) {
    const v16bf bb = __builtin_convertvector(*(const v16f*)(brow + k0), v16bf);

    // Stage all 8 A tiles into distinct registers.
    AU a[GTiles];
    #pragma unroll
    for (int g8 = 0; g8 < GTiles; ++g8) {
      const float* ar = arow0 + (size_t)g8 * 16 * C + k0;
      a[g8].h[0] = __builtin_convertvector(*(const v8f*)(ar),      v8bf);
      a[g8].h[1] = __builtin_convertvector(*(const v8f*)(ar + 8), v8bf);
    }

    // Back-to-back independent WMMA burst (8 accumulators).
    #pragma unroll
    for (int g8 = 0; g8 < GTiles; ++g8) {
      // (neg_a, A, neg_b, B, c_mod, C, reuse_a, reuse_b)
      acc[g8] = __builtin_amdgcn_wmma_f32_16x16x32_bf16(false, a[g8].v, false,
                                                        bb, (short)0, acc[g8],
                                                        false, false);
    }
  }

  // D layout: VGPR r -> (M = r + half*8, N = m)
  const float rqv = rq[(size_t)b * Q + qt * 16 + m];
  const int moff = half * 8;
  #pragma unroll
  for (int g8 = 0; g8 < GTiles; ++g8) {
    float* crow = cost + ((size_t)b * G + (size_t)g8 * 16) * Q + (size_t)qt * 16;
    #pragma unroll
    for (int r = 0; r < 8; ++r) {
      const float rgv = rg[(size_t)b * G + g8 * 16 + r + moff];
      crow[(size_t)(r + moff) * Q + m] = 1.0f - acc[g8][r] * rqv * rgv;
    }
  }
}

// ---------------------------------------------------------------------------
// Kernel 3: Hungarian (shortest augmenting path LAP), one block per batch.
// Exact port of the reference algorithm; cost is [G, Q] with G=128 <= Q=2048.
// All per-problem state lives in LDS (~35 KB).  Column argmin is a block-wide
// reduction with lowest-index tie-break (matches np.argmin).
// ---------------------------------------------------------------------------
__global__ __launch_bounds__(256)
void hungarian_kernel(const float* __restrict__ costAll,
                      int* __restrict__ matched) {
  const int b   = blockIdx.x;
  const int tid = threadIdx.x;
  const float* cost = costAll + (size_t)b * G * Q;

  __shared__ float v_[Q];
  __shared__ float spc[Q];
  __shared__ int   path[Q];
  __shared__ int   row4col[Q];
  __shared__ unsigned char SC[Q];
  __shared__ float u_[G];
  __shared__ unsigned char SR[G];
  __shared__ int   col4row[G];
  __shared__ float redV[256];
  __shared__ int   redI[256];
  __shared__ int   s_i, s_sink;
  __shared__ float s_min;

  for (int j = tid; j < Q; j += 256) { v_[j] = 0.f; row4col[j] = -1; }
  for (int k = tid; k < G; k += 256) { u_[k] = 0.f; col4row[k] = -1; }
  __syncthreads();

  for (int cur = 0; cur < G; ++cur) {
    for (int j = tid; j < Q; j += 256) { spc[j] = INF; SC[j] = 0; }
    for (int k = tid; k < G; k += 256) SR[k] = 0;
    if (tid == 0) { s_i = cur; s_min = 0.f; s_sink = -1; }
    __syncthreads();

    while (true) {
      const int   i      = s_i;
      const float minVal = s_min;
      if (tid == 0) SR[i] = 1;
      const float ui = u_[i];
      const float* crow = cost + (size_t)i * Q;

      // Relax all unvisited columns; track this thread's best (value, index).
      float bestv = INF;
      int   bestj = Q;
      for (int j = tid; j < Q; j += 256) {
        if (!SC[j]) {
          float r = minVal + crow[j] - ui - v_[j];
          if (r < spc[j]) { spc[j] = r; path[j] = i; }
          float sv = spc[j];
          if (sv < bestv) { bestv = sv; bestj = j; }
        }
      }
      redV[tid] = bestv; redI[tid] = bestj;
      __syncthreads();
      for (int off = 128; off > 0; off >>= 1) {
        if (tid < off) {
          float ov = redV[tid + off]; int oj = redI[tid + off];
          if (ov < redV[tid] || (ov == redV[tid] && oj < redI[tid])) {
            redV[tid] = ov; redI[tid] = oj;
          }
        }
        __syncthreads();
      }
      if (tid == 0) {
        int j = redI[0];
        s_min = redV[0];
        SC[j] = 1;
        if (row4col[j] < 0) s_sink = j;
        else                s_i    = row4col[j];
      }
      __syncthreads();
      if (s_sink >= 0) break;
    }

    // Dual updates.
    const float minVal = s_min;
    for (int k = tid; k < G; k += 256)
      if (SR[k] && k != cur) u_[k] += minVal - spc[col4row[k]];
    for (int j = tid; j < Q; j += 256)
      if (SC[j]) v_[j] -= minVal - spc[j];
    if (tid == 0) u_[cur] += minVal;
    __syncthreads();

    // Augment along the alternating path (serial, <= G steps).
    if (tid == 0) {
      int j = s_sink;
      while (true) {
        int i = path[j];
        row4col[j] = i;
        int nj = col4row[i];
        col4row[i] = j;
        j = nj;
        if (i == cur) break;
      }
    }
    __syncthreads();
  }

  for (int k = tid; k < G; k += 256) matched[b * G + k] = col4row[k];
}

// ---------------------------------------------------------------------------
// Kernel 4: gather outputs. d_out = [seed_signatures (B*G*C) | mask (B*G) |
// seed_scores (B*G)] as float32.
// ---------------------------------------------------------------------------
__global__ void gather_kernel(const float* __restrict__ qsig,
                              const float* __restrict__ qlogits,
                              const unsigned char* __restrict__ gmask,
                              const int* __restrict__ matched,
                              float* __restrict__ out) {
  const int bg = blockIdx.x;           // 0 .. B*G-1
  const int b  = bg / G;
  const int qi = matched[bg];
  const int mk = gmask[bg] ? 1 : 0;
  const float* src = qsig + ((size_t)b * Q + qi) * C;
  float* dsig = out + (size_t)bg * C;
  for (int i = threadIdx.x; i < C; i += blockDim.x)
    dsig[i] = mk ? src[i] : 0.f;
  if (threadIdx.x == 0) {
    float* dmask  = out + (size_t)B * G * C;
    float* dscore = dmask + (size_t)B * G;
    dmask[bg] = (float)mk;
    float lg = qlogits[(size_t)b * Q + qi];
    dscore[bg] = mk ? (1.f / (1.f + expf(-lg))) : 0.f;
  }
}

// ---------------------------------------------------------------------------
extern "C" void kernel_launch(void* const* d_in, const int* in_sizes, int n_in,
                              void* d_out, int out_size, void* d_ws, size_t ws_size,
                              hipStream_t stream) {
  const float*         qsig  = (const float*)d_in[0];          // [B,Q,C]
  const float*         qlog  = (const float*)d_in[1];          // [B,Q]
  const float*         gsig  = (const float*)d_in[2];          // [B,G,C]
  const unsigned char* gmask = (const unsigned char*)d_in[3];  // [B,G] bool
  float* out = (float*)d_out;

  // Workspace layout (~8.5 MB total)
  char* ws = (char*)d_ws;
  float* rq      = (float*)ws;                                   // B*Q floats
  float* rg      = (float*)(ws + (size_t)B * Q * 4);             // B*G floats
  float* cost    = (float*)(ws + (size_t)B * Q * 4 + (size_t)B * G * 4);
  int*   matched = (int*)  (ws + (size_t)B * Q * 4 + (size_t)B * G * 4
                               + (size_t)B * G * Q * 4);

  rnorm_kernel<<<B * Q, 256, 0, stream>>>(qsig, rq);
  rnorm_kernel<<<B * G, 256, 0, stream>>>(gsig, rg);

  constexpr int jobs = B * (Q / 16);   // 1024 waves, 8 waves/block
  cost_gemm_kernel<<<jobs / 8, 256, 0, stream>>>(qsig, gsig, rq, rg, cost);

  hungarian_kernel<<<B, 256, 0, stream>>>(cost, matched);

  gather_kernel<<<B * G, 256, 0, stream>>>(qsig, qlog, gmask, matched, out);
}
